// FFLayer_14456859918406
// MI455X (gfx1250) — compile-verified, matrix operand order checked by reading, not run on hardware
//
#include <hip/hip_runtime.h>

// ---------------------------------------------------------------------------
// out = relu( (x / (||x||_2 + 1e-4)) @ W^T + b )
// x: [131072, 1024] f32, W: [256, 1024] f32, b: [256] f32 -> out [131072,256]
//
// Normalization commutes with the matmul (per-row scalar), so we do:
//   out = relu( (x @ W^T) * inv_norm_row + b )
// Matmul in bf16 WMMA (16x16x32, f32 accum) -> memory-bound at ~23.3 TB/s.
// Single pass over x: sum-of-squares computed while converting x->bf16 in LDS.
// ---------------------------------------------------------------------------

typedef __attribute__((ext_vector_type(16))) __bf16 v16bf;
typedef __attribute__((ext_vector_type(8)))  float  v8f;

#define IN_F   1024
#define OUT_F  256
#define BATCH  131072
#define BM     32           // rows per block (64 KB LDS for bf16 x-tile)
#define EPS    1e-4f

__device__ __forceinline__ unsigned f32_to_bf16_rne(float f) {
    unsigned u = __float_as_uint(f);
    return (u + 0x7fffu + ((u >> 16) & 1u)) >> 16;   // round-to-nearest-even
}

// --- pre-pass: W f32 -> bf16 into workspace (512 KB, L2-resident) ----------
__global__ __launch_bounds__(256) void convert_w_kernel(
        const float* __restrict__ W, unsigned short* __restrict__ Wbf) {
    int i = blockIdx.x * 256 + threadIdx.x;          // 65536 threads x 4 elems
    float4 w = ((const float4*)W)[i];
    unsigned lo = f32_to_bf16_rne(w.x) | (f32_to_bf16_rne(w.y) << 16);
    unsigned hi = f32_to_bf16_rne(w.z) | (f32_to_bf16_rne(w.w) << 16);
    ((uint2*)Wbf)[i] = make_uint2(lo, hi);
}

// --- main fused kernel -----------------------------------------------------
__global__ __launch_bounds__(256) void ff_norm_gemm_relu_kernel(
        const float* __restrict__ X,
        const unsigned short* __restrict__ Wbf,
        const float* __restrict__ bias,
        float* __restrict__ out) {
    __shared__ unsigned short lds_x[BM][IN_F];       // 64 KB bf16 x-tile
    __shared__ float lds_inv[BM];                    // per-row 1/(||x||+eps)

    const int tid      = threadIdx.x;
    const int row_base = blockIdx.x * BM;

    if (tid < BM) lds_inv[tid] = 0.0f;
    __syncthreads();

    // ---- phase 1: stream rows, sum-of-squares + bf16 convert into LDS ----
    {
        const int r = tid >> 3;                      // 0..31  (row in tile)
        const int q = tid & 7;                       // 8 threads per row
        const float4* src = (const float4*)(X + (size_t)(row_base + r) * IN_F + q * 128);
        uint2* dst = (uint2*)(&lds_x[r][q * 128]);
        float ss = 0.0f;
        #pragma unroll 8
        for (int i = 0; i < 32; ++i) {               // 128 floats per thread
            float4 v = src[i];
            ss += v.x * v.x + v.y * v.y + v.z * v.z + v.w * v.w;
            unsigned lo = f32_to_bf16_rne(v.x) | (f32_to_bf16_rne(v.y) << 16);
            unsigned hi = f32_to_bf16_rne(v.z) | (f32_to_bf16_rne(v.w) << 16);
            dst[i] = make_uint2(lo, hi);
        }
        atomicAdd(&lds_inv[r], ss);                  // ds_add_f32
    }
    __syncthreads();
    if (tid < BM) lds_inv[tid] = 1.0f / (sqrtf(lds_inv[tid]) + EPS);
    __syncthreads();

    // ---- phase 2: bf16 WMMA matmul -----------------------------------
    // 8 waves: wm in {0,1} picks 16-row slice, wn in {0..3} picks 64-col slice.
    const int wave = tid >> 5;
    const int lane = tid & 31;
    const int half = lane >> 4;                      // 0: lanes 0-15, 1: 16-31
    const int lrow = lane & 15;
    const int wm   = wave & 1;
    const int wn   = wave >> 1;

    const int arow = wm * 16 + lrow;                 // A-fragment source row
    const int nb0  = wn * 64;                        // wave column base

    v8f c[4];
    #pragma unroll
    for (int t = 0; t < 4; ++t) c[t] = (v8f){0,0,0,0,0,0,0,0};

    for (int kb = 0; kb < IN_F; kb += 32) {
        // ISA 16-bit A/B layout: lanes 0-15 hold K{+0..7, +16..23},
        // lanes 16-31 hold K{+8..15, +24..31} -> two 16-byte chunks per lane.
        const int k0 = kb + half * 8;
        const int k1 = k0 + 16;

        union { uint4 u[2]; v16bf v; } a;
        a.u[0] = *(const uint4*)(&lds_x[arow][k0]);
        a.u[1] = *(const uint4*)(&lds_x[arow][k1]);

        #pragma unroll
        for (int t = 0; t < 4; ++t) {
            // B[k][n] = W[nb0 + t*16 + lrow][k]  (W rows give B columns)
            const unsigned short* wp = Wbf + (size_t)(nb0 + t * 16 + lrow) * IN_F;
            union { uint4 u[2]; v16bf v; } bm;
            bm.u[0] = *(const uint4*)(wp + k0);
            bm.u[1] = *(const uint4*)(wp + k1);
            c[t] = __builtin_amdgcn_wmma_f32_16x16x32_bf16(
                       false, a.v, false, bm.v, (short)0, c[t], false, false);
        }
    }

    // ---- epilogue: scale by inv-norm, +bias, ReLU, store f32 -------------
    // C layout: lane half 0 -> M = v, half 1 -> M = v + 8; col = lane % 16.
    #pragma unroll
    for (int t = 0; t < 4; ++t) {
        const int col = nb0 + t * 16 + lrow;
        const float bb = bias[col];
        #pragma unroll
        for (int v = 0; v < 8; ++v) {
            const int lr = wm * 16 + v + half * 8;   // row within block tile
            float val = c[t][v] * lds_inv[lr] + bb;
            val = val > 0.0f ? val : 0.0f;
            out[(size_t)(row_base + lr) * OUT_F + col] = val;
        }
    }
}

extern "C" void kernel_launch(void* const* d_in, const int* in_sizes, int n_in,
                              void* d_out, int out_size, void* d_ws, size_t ws_size,
                              hipStream_t stream) {
    const float* X  = (const float*)d_in[0];   // [131072,1024]
    const float* W  = (const float*)d_in[1];   // [256,1024]
    const float* Bv = (const float*)d_in[2];   // [256]
    float* out = (float*)d_out;
    unsigned short* Wbf = (unsigned short*)d_ws;   // 512 KB bf16 W

    // convert W once per launch (deterministic, cheap, L2-resident afterwards)
    convert_w_kernel<<<256, 256, 0, stream>>>(W, Wbf);

    ff_norm_gemm_relu_kernel<<<BATCH / BM, 256, 0, stream>>>(X, Wbf, Bv, out);
}
